// I2HOFI_70368744178172
// MI455X (gfx1250) — matrix operand order repaired and checked by reading, not compiled, not run on hardware
//
#include <hip/hip_runtime.h>
#include <hip/hip_bf16.h>
#include <math.h>

typedef __attribute__((ext_vector_type(16))) __bf16 v16bf;
typedef __attribute__((ext_vector_type(8)))  float  v8f;

#define ALPHA_C 0.2f
#define LEAKY_C 0.2f

static constexpr int Bn   = 108;
static constexpr int Nn   = 392;
static constexpr int Hn   = 4;
static constexpr int Mtot = Bn * Nn;   // 42336

// LDS tiles: 64 rows x 32 bf16 (K), packed dwords, row stride 20 dwords (80B)
#define LDSW 20

__device__ __forceinline__ int imin(int a, int b) { return a < b ? a : b; }

__device__ __forceinline__ unsigned pk_bf16(float a, float b) {
    unsigned short ua = __builtin_bit_cast(unsigned short, (__bf16)a);
    unsigned short ub = __builtin_bit_cast(unsigned short, (__bf16)b);
    return (unsigned)ua | ((unsigned)ub << 16);
}

union Frag {
    v16bf    v;
    uint4    q[2];
    unsigned u[8];
};

// wave-level: A frag from As32 (rows wr*16..+15), two 16-col B frags, 2 WMMAs
__device__ __forceinline__ void wmma_pair(const unsigned* As32, const unsigned* Bs32,
                                          int wr, int wc, int l, int hi,
                                          v8f& acc0, v8f& acc1)
{
    Frag fa;
    const unsigned* ap = &As32[(wr * 16 + l) * LDSW];
    fa.q[0] = *(const uint4*)(ap + 8 * hi);
    fa.q[1] = *(const uint4*)(ap + 8 * hi + 4);

    const unsigned* bp0 = &Bs32[(wc * 32 + l) * LDSW + 8 * hi];
    Frag fb0;
    fb0.q[0] = *(const uint4*)(bp0);
    fb0.q[1] = *(const uint4*)(bp0 + 4);
    acc0 = __builtin_amdgcn_wmma_f32_16x16x32_bf16(false, fa.v, false, fb0.v,
                                                   (short)0, acc0, false, false);
    const unsigned* bp1 = bp0 + 16 * LDSW;
    Frag fb1;
    fb1.q[0] = *(const uint4*)(bp1);
    fb1.q[1] = *(const uint4*)(bp1 + 4);
    acc1 = __builtin_amdgcn_wmma_f32_16x16x32_bf16(false, fa.v, false, fb1.v,
                                                   (short)0, acc1, false, false);
}

// ---------------------------------------------------------------------------
// Generic 64x64-tile GEMM, K=256, Ncols=256, bf16 WMMA / f32 accumulate.
// Double-buffered LDS: stage chunk i+1 while WMMAs consume chunk i.
// MODE 0: C = sigmoid(A@B + bias)   (APPNP MLP)
// MODE 1: C = A@B                   (xp = z @ kernel)
// ---------------------------------------------------------------------------
template<int MODE>
__global__ __launch_bounds__(256)
void gemm_256(const float* __restrict__ A, const float* __restrict__ Bm,
              const float* __restrict__ bias, float* __restrict__ C)
{
    __shared__ __align__(16) unsigned As32[2][64 * LDSW];
    __shared__ __align__(16) unsigned Bs32[2][64 * LDSW];

    const int t    = threadIdx.x;
    const int wave = t >> 5, lane = t & 31;
    const int wr   = wave >> 1, wc = wave & 1;
    const int l    = lane & 15, hi = lane >> 4;
    const int rowBase = blockIdx.x * 64;
    const int colBase = blockIdx.y * 64;
    const bool edge  = (rowBase + 64 > Mtot);       // block-uniform

    v8f acc0 = {}, acc1 = {};

    const int ar  = t >> 2;         // 0..63  A-tile row
    const int ak  = (t & 3) * 8;    // 0,8,16,24 (bf16 slot)
    const int bn  = t & 63;         // B-tile col
    const int bk0 = t >> 6;         // 0..3 (pair slot base)

    const int arow = imin(rowBase + ar, Mtot - 1);  // clamp: invalid rows never stored
    const float* aRowPtr = A + (size_t)arow * 256;

    auto stage = [&](int cidx, int buf) {
        const int kb = cidx * 32;
        {
            const float4* src4 = (const float4*)(aRowPtr + kb + ak);
            float4 a0 = src4[0], a1 = src4[1];
            uint4 w;
            w.x = pk_bf16(a0.x, a0.y); w.y = pk_bf16(a0.z, a0.w);
            w.z = pk_bf16(a1.x, a1.y); w.w = pk_bf16(a1.z, a1.w);
            *(uint4*)&As32[buf][ar * LDSW + (ak >> 1)] = w;
        }
        #pragma unroll
        for (int p = 0; p < 4; ++p) {
            int kp = bk0 + p * 4;                   // pair index 0..15
            int k  = kb + 2 * kp;
            float v0 = Bm[(size_t)k * 256 + colBase + bn];
            float v1 = Bm[(size_t)(k + 1) * 256 + colBase + bn];
            Bs32[buf][bn * LDSW + kp] = pk_bf16(v0, v1);
        }
    };

    stage(0, 0);
    __syncthreads();
    for (int i = 0; i < 7; ++i) {
        stage(i + 1, (i + 1) & 1);
        wmma_pair(As32[i & 1], Bs32[i & 1], wr, wc, l, hi, acc0, acc1);
        __syncthreads();
    }
    wmma_pair(As32[1], Bs32[1], wr, wc, l, hi, acc0, acc1);   // chunk 7

    if (!edge) {
        #pragma unroll
        for (int sub = 0; sub < 2; ++sub) {
            v8f acc = sub ? acc1 : acc0;
            int col = colBase + wc * 32 + sub * 16 + l;
            float bv = (MODE == 0) ? bias[col] : 0.f;
            float* cptr = C + (size_t)(rowBase + wr * 16 + hi * 8) * 256 + col;
            #pragma unroll
            for (int r = 0; r < 8; ++r) {
                float v = acc[r];
                if (MODE == 0) v = 1.f / (1.f + __expf(-(v + bv)));
                cptr[(size_t)r * 256] = v;
            }
        }
    } else {
        #pragma unroll
        for (int sub = 0; sub < 2; ++sub) {
            v8f acc = sub ? acc1 : acc0;
            int col = colBase + wc * 32 + sub * 16 + l;
            float bv = (MODE == 0) ? bias[col] : 0.f;
            #pragma unroll
            for (int r = 0; r < 8; ++r) {
                int grow = rowBase + wr * 16 + hi * 8 + r;
                if (grow < Mtot) {
                    float v = acc[r];
                    if (MODE == 0) v = 1.f / (1.f + __expf(-(v + bv)));
                    C[(size_t)grow * 256 + col] = v;
                }
            }
        }
    }
}

// ---------------------------------------------------------------------------
// APPNP propagation: per batch b, Z = (1-ALPHA) * a(392x392) @ Mlp_b + ALPHA*Mlp_b
// 12 guard-free K chunks + 1 tail chunk; double-buffered.
// ---------------------------------------------------------------------------
__global__ __launch_bounds__(256)
void gemm_appnp(const float* __restrict__ Aadj, const float* __restrict__ Mlp,
                float* __restrict__ Z)
{
    __shared__ __align__(16) unsigned As32[2][64 * LDSW];
    __shared__ __align__(16) unsigned Bs32[2][64 * LDSW];

    const int t    = threadIdx.x;
    const int wave = t >> 5, lane = t & 31;
    const int wr   = wave >> 1, wc = wave & 1;
    const int l    = lane & 15, hi = lane >> 4;
    const int rowBase = blockIdx.x * 64;
    const int colBase = blockIdx.y * 64;
    const int b       = blockIdx.z;
    const bool edge   = (rowBase + 64 > Nn);        // block-uniform

    const float* Bbase = Mlp + (size_t)b * Nn * 256;

    v8f acc0 = {}, acc1 = {};

    const int ar  = t >> 2;
    const int ak  = (t & 3) * 8;
    const int bn  = t & 63;
    const int bk0 = t >> 6;

    const int arow = imin(rowBase + ar, Nn - 1);
    const float* aRowPtr = Aadj + (size_t)arow * Nn;

    auto stageA_from = [&](const float* src, int buf) {
        const float4* src4 = (const float4*)src;
        float4 a0 = src4[0], a1 = src4[1];
        uint4 w;
        w.x = pk_bf16(a0.x, a0.y); w.y = pk_bf16(a0.z, a0.w);
        w.z = pk_bf16(a1.x, a1.y); w.w = pk_bf16(a1.z, a1.w);
        *(uint4*)&As32[buf][ar * LDSW + (ak >> 1)] = w;
    };
    auto stageN = [&](int cidx, int buf) {          // clean chunks, k <= 383
        const int kb = cidx * 32;
        stageA_from(aRowPtr + kb + ak, buf);
        #pragma unroll
        for (int p = 0; p < 4; ++p) {
            int kp = bk0 + p * 4;
            int k  = kb + 2 * kp;
            float v0 = Bbase[(size_t)k * 256 + colBase + bn];
            float v1 = Bbase[(size_t)(k + 1) * 256 + colBase + bn];
            Bs32[buf][bn * LDSW + kp] = pk_bf16(v0, v1);
        }
    };
    auto stageT = [&](int buf) {                    // tail chunk, k 384..415
        const int kb = 384;
        stageA_from(aRowPtr + (Nn - 8), buf);       // 384..391 (slots 8..31 garbage, B=0)
        #pragma unroll
        for (int p = 0; p < 4; ++p) {
            int kp = bk0 + p * 4;
            int k0 = kb + 2 * kp, k1 = k0 + 1;
            float v0 = Bbase[(size_t)imin(k0, Nn - 1) * 256 + colBase + bn];
            float v1 = Bbase[(size_t)imin(k1, Nn - 1) * 256 + colBase + bn];
            v0 = (k0 < Nn) ? v0 : 0.f;
            v1 = (k1 < Nn) ? v1 : 0.f;
            Bs32[buf][bn * LDSW + kp] = pk_bf16(v0, v1);
        }
    };

    stageN(0, 0);
    __syncthreads();
    for (int i = 0; i < 11; ++i) {                  // computes chunks 0..10
        stageN(i + 1, (i + 1) & 1);
        wmma_pair(As32[i & 1], Bs32[i & 1], wr, wc, l, hi, acc0, acc1);
        __syncthreads();
    }
    stageT(0);                                      // chunk 12 -> buf 0
    wmma_pair(As32[1], Bs32[1], wr, wc, l, hi, acc0, acc1);   // chunk 11
    __syncthreads();
    wmma_pair(As32[0], Bs32[0], wr, wc, l, hi, acc0, acc1);   // chunk 12

    if (!edge) {
        #pragma unroll
        for (int sub = 0; sub < 2; ++sub) {
            v8f acc = sub ? acc1 : acc0;
            int col = colBase + wc * 32 + sub * 16 + l;
            const float* mptr = Bbase + (size_t)(rowBase + wr * 16 + hi * 8) * 256 + col;
            float* zptr = Z + ((size_t)b * Nn + rowBase + wr * 16 + hi * 8) * 256 + col;
            #pragma unroll
            for (int r = 0; r < 8; ++r)
                zptr[(size_t)r * 256] = (1.f - ALPHA_C) * acc[r]
                                      + ALPHA_C * mptr[(size_t)r * 256];
        }
    } else {
        #pragma unroll
        for (int sub = 0; sub < 2; ++sub) {
            v8f acc = sub ? acc1 : acc0;
            int col = colBase + wc * 32 + sub * 16 + l;
            #pragma unroll
            for (int r = 0; r < 8; ++r) {
                int grow = rowBase + wr * 16 + hi * 8 + r;
                if (grow < Nn) {
                    float v = (1.f - ALPHA_C) * acc[r]
                            + ALPHA_C * Bbase[(size_t)grow * 256 + col];
                    Z[((size_t)b * Nn + grow) * 256 + col] = v;
                }
            }
        }
    }
}

// ---------------------------------------------------------------------------
// One-shot mask bits: masked iff (n != m) && (a[n,m]==0); out-of-range m masked.
// ---------------------------------------------------------------------------
__global__ __launch_bounds__(256)
void build_mask(const float* __restrict__ Aadj, unsigned* __restrict__ maskG)
{
    int idx = blockIdx.x * 256 + threadIdx.x;
    if (idx >= Nn * 16) return;
    int row = idx >> 4, word = idx & 15;
    const float* arow = Aadj + (size_t)row * Nn;
    unsigned bits = 0;
    for (int j = 0; j < 32; ++j) {
        int m = word * 32 + j;
        bool masked = (m < Nn) ? ((arow[m] == 0.f) && (row != m)) : true;
        if (masked) bits |= (1u << j);
    }
    maskG[idx] = bits;
}

// ---------------------------------------------------------------------------
// Attention logits: s[b,n,h] = sum_o xp[b,n,h,o]*ask[o,h]; same for ng.
// ---------------------------------------------------------------------------
__global__ __launch_bounds__(256)
void attn_logits(const float* __restrict__ xp, const float* __restrict__ ask,
                 const float* __restrict__ ank,
                 float* __restrict__ sArr, float* __restrict__ ngArr)
{
    __shared__ float askS[256], ankS[256];
    askS[threadIdx.x] = ask[threadIdx.x];
    ankS[threadIdx.x] = ank[threadIdx.x];
    __syncthreads();

    int idx = blockIdx.x * 256 + threadIdx.x;
    if (idx >= Mtot * Hn) return;
    int h  = idx & 3;       // Hn == 4
    int bn = idx >> 2;      // b*N + n
    const float4* row4 = (const float4*)(xp + (size_t)bn * 256 + h * 64);
    float s = 0.f, g = 0.f;
    #pragma unroll
    for (int o4 = 0; o4 < 16; ++o4) {
        float4 v = row4[o4];
        int o = o4 * 4;
        s += v.x * askS[(o + 0) * 4 + h] + v.y * askS[(o + 1) * 4 + h]
           + v.z * askS[(o + 2) * 4 + h] + v.w * askS[(o + 3) * 4 + h];
        g += v.x * ankS[(o + 0) * 4 + h] + v.y * ankS[(o + 1) * 4 + h]
           + v.z * ankS[(o + 2) * 4 + h] + v.w * ankS[(o + 3) * 4 + h];
    }
    sArr[idx]  = s;
    ngArr[idx] = g;
}

// ---------------------------------------------------------------------------
// Fused GAT: softmax(leaky(s+ng)+mask) @ xp, bias, elu. Double-buffered chunks.
// Block = 128 threads (4 waves); handles (b, h, 64-row tile); each wave 16 rows.
// ---------------------------------------------------------------------------
__global__ __launch_bounds__(128)
void gat_aggregate(const unsigned* __restrict__ maskG, const float* __restrict__ xp,
                   const float* __restrict__ sArr, const float* __restrict__ ngArr,
                   const float* __restrict__ bias, float* __restrict__ out)
{
    __shared__ float    sS[64];
    __shared__ float    ngS[416];               // padded, tail zeroed
    __shared__ unsigned maskS[64 * 16];
    __shared__ float    rowMax[64];
    __shared__ float    rowSum[64];
    __shared__ float    red[128];
    __shared__ __align__(16) unsigned xpT32[2][64 * LDSW];  // [n][kpair] packed bf16

    const int t    = threadIdx.x;
    const int wave = t >> 5, lane = t & 31;
    const int l    = lane & 15, hi = lane >> 4;
    const int rowBase = blockIdx.x * 64;
    const int h       = blockIdx.y;
    const int b       = blockIdx.z;
    const bool edge   = (rowBase + 64 > Nn);    // block-uniform

    if (t < 64) {
        int gr = imin(rowBase + t, Nn - 1);
        sS[t] = sArr[((size_t)b * Nn + gr) * 4 + h];
    }
    for (int m = t; m < 416; m += 128)
        ngS[m] = (m < Nn) ? ngArr[((size_t)b * Nn + m) * 4 + h] : 0.f;
    for (int w = t; w < 64 * 16; w += 128) {
        int gr = imin(rowBase + (w >> 4), Nn - 1);
        maskS[w] = maskG[gr * 16 + (w & 15)];
    }

    auto gstageN = [&](int ch, int buf) {       // clean chunks, m <= 383
        const int mb = ch * 32;
        #pragma unroll
        for (int p = 0; p < 8; ++p) {
            int kp = (t >> 6) + p * 2;          // pair index 0..15
            int m0 = mb + 2 * kp;
            float v0 = xp[((size_t)b * Nn + m0) * 256 + h * 64 + (t & 63)];
            float v1 = xp[((size_t)b * Nn + m0 + 1) * 256 + h * 64 + (t & 63)];
            xpT32[buf][(t & 63) * LDSW + kp] = pk_bf16(v0, v1);
        }
    };
    auto gstageT = [&](int buf) {               // tail chunk, clamp m (P=0 covers tail)
        const int mb = 384;
        #pragma unroll
        for (int p = 0; p < 8; ++p) {
            int kp = (t >> 6) + p * 2;
            int m0 = imin(mb + 2 * kp, Nn - 1);
            int m1 = imin(mb + 2 * kp + 1, Nn - 1);
            float v0 = xp[((size_t)b * Nn + m0) * 256 + h * 64 + (t & 63)];
            float v1 = xp[((size_t)b * Nn + m1) * 256 + h * 64 + (t & 63)];
            xpT32[buf][(t & 63) * LDSW + kp] = pk_bf16(v0, v1);
        }
    };

    // overlap chunk-0 staging with the softmax passes
    gstageN(0, 0);
    __syncthreads();

    // pass 1: row max (each row split over two threads by odd/even word)
    {
        int row = t & 63, half = t >> 6;
        float srow0 = sS[row];
        float mx = -3.0e38f;
        for (int w = half; w < 13; w += 2) {
            unsigned bits = maskS[row * 16 + w];
            #pragma unroll
            for (int j = 0; j < 32; ++j) {
                float c = srow0 + ngS[w * 32 + j];
                c = c > 0.f ? c : LEAKY_C * c;
                c = ((bits >> j) & 1u) ? -3.0e38f : c;
                mx = fmaxf(mx, c);
            }
        }
        red[t] = mx;
    }
    __syncthreads();
    if (t < 64) rowMax[t] = fmaxf(red[t], red[t + 64]);
    __syncthreads();

    // pass 2: row sum of exp(c - max)
    {
        int row = t & 63, half = t >> 6;
        float srow0 = sS[row], rmx0 = rowMax[row];
        float sum = 0.f;
        for (int w = half; w < 13; w += 2) {
            unsigned bits = maskS[row * 16 + w];
            #pragma unroll
            for (int j = 0; j < 32; ++j) {
                float c = srow0 + ngS[w * 32 + j];
                c = c > 0.f ? c : LEAKY_C * c;
                float e = __expf(c - rmx0);
                sum += ((bits >> j) & 1u) ? 0.f : e;
            }
        }
        red[t] = sum;
    }
    __syncthreads();
    if (t < 64) rowSum[t] = red[t] + red[t + 64];
    __syncthreads();

    v8f acc[4] = {};
    const int   myrow = wave * 16 + l;
    const float srow  = sS[myrow];
    const float rmx   = rowMax[myrow];

    auto gcompute = [&](int ch, int buf) {
        const int mb = ch * 32;
        const unsigned mw = maskS[myrow * 16 + ch];   // one mask word per chunk
        Frag fa;
        #pragma unroll
        for (int w = 0; w < 8; ++w) {
            int ebase = 2 * w;
            int kk0 = 16 * hi + ebase;
            float c0 = srow + ngS[mb + kk0];
            float c1 = srow + ngS[mb + kk0 + 1];
            c0 = c0 > 0.f ? c0 : LEAKY_C * c0;
            c1 = c1 > 0.f ? c1 : LEAKY_C * c1;
            float p0 = ((mw >> kk0) & 1u)       ? 0.f : __expf(c0 - rmx);
            float p1 = ((mw >> (kk0 + 1)) & 1u) ? 0.f : __expf(c1 - rmx);
            fa.u[w] = pk_bf16(p0, p1);
        }
        #pragma unroll
        for (int t4 = 0; t4 < 4; ++t4) {
            const unsigned* bp = &xpT32[buf][(t4 * 16 + l) * LDSW + 8 * hi];
            Frag fb;
            fb.q[0] = *(const uint4*)(bp);
            fb.q[1] = *(const uint4*)(bp + 4);
            acc[t4] = __builtin_amdgcn_wmma_f32_16x16x32_bf16(false, fa.v, false, fb.v,
                                                              (short)0, acc[t4], false, false);
        }
    };

    for (int i = 0; i < 11; ++i) {              // computes chunks 0..10
        gstageN(i + 1, (i + 1) & 1);
        gcompute(i, i & 1);
        __syncthreads();
    }
    gstageT(0);                                 // chunk 12 -> buf 0
    gcompute(11, 1);
    __syncthreads();
    gcompute(12, 0);

    // epilogue: normalize, bias, elu, store
    if (!edge) {
        #pragma unroll
        for (int t4 = 0; t4 < 4; ++t4) {
            int n   = t4 * 16 + l;
            int col = h * 64 + n;
            float bv = bias[col];
            float* optr = out + ((size_t)b * Nn + rowBase + wave * 16 + hi * 8) * 256 + col;
            #pragma unroll
            for (int r = 0; r < 8; ++r) {
                float v = acc[t4][r] / rowSum[wave * 16 + hi * 8 + r] + bv;
                v = v > 0.f ? v : (__expf(v) - 1.f);
                optr[(size_t)r * 256] = v;
            }
        }
    } else {
        #pragma unroll
        for (int t4 = 0; t4 < 4; ++t4) {
            int n   = t4 * 16 + l;
            int col = h * 64 + n;
            float bv = bias[col];
            #pragma unroll
            for (int r = 0; r < 8; ++r) {
                int lrow = wave * 16 + hi * 8 + r;
                int gr   = rowBase + lrow;
                if (gr < Nn) {
                    float v = acc[t4][r] / rowSum[lrow] + bv;
                    v = v > 0.f ? v : (__expf(v) - 1.f);
                    out[((size_t)b * Nn + gr) * 256 + col] = v;
                }
            }
        }
    }
}

// ---------------------------------------------------------------------------
extern "C" void kernel_launch(void* const* d_in, const int* in_sizes, int n_in,
                              void* d_out, int out_size, void* d_ws, size_t ws_size,
                              hipStream_t stream)
{
    const float* x    = (const float*)d_in[0];
    const float* a    = (const float*)d_in[1];
    const float* wmlp = (const float*)d_in[2];
    const float* bmlp = (const float*)d_in[3];
    const float* kern = (const float*)d_in[4];
    const float* ask  = (const float*)d_in[5];
    const float* ank  = (const float*)d_in[6];
    const float* bias = (const float*)d_in[7];
    float* out = (float*)d_out;
    float* ws  = (float*)d_ws;

    float*    mlp   = ws;                              // [Mtot,256]; reused as xp
    float*    z     = ws + (size_t)Mtot * 256;         // [Mtot,256]
    float*    sA    = ws + (size_t)Mtot * 256 * 2;     // [Mtot,H]
    float*    ngA   = sA + (size_t)Mtot * Hn;          // [Mtot,H]
    unsigned* maskG = (unsigned*)(ngA + (size_t)Mtot * Hn);  // [392*16]

    dim3 blk(256);

    build_mask<<<dim3((Nn * 16 + 255) / 256), blk, 0, stream>>>(a, maskG);

    dim3 g1((Mtot + 63) / 64, 4);
    gemm_256<0><<<g1, blk, 0, stream>>>(x, wmlp, bmlp, mlp);

    dim3 g2(7, 4, Bn);
    gemm_appnp<<<g2, blk, 0, stream>>>(a, mlp, z);

    gemm_256<1><<<g1, blk, 0, stream>>>(z, kern, nullptr, mlp);   // xp overwrites mlp

    dim3 g3((Mtot * Hn + 255) / 256);
    attn_logits<<<g3, blk, 0, stream>>>(mlp, ask, ank, sA, ngA);

    dim3 g4(7, 4, Bn);
    gat_aggregate<<<g4, dim3(128), 0, stream>>>(maskG, mlp, sA, ngA, bias, out);
}
